// SSIMLoss_21964462752188
// MI455X (gfx1250) — matrix operand order repaired and checked by reading, not compiled, not run on hardware
//
#include <hip/hip_runtime.h>
#include <math.h>

typedef _Float16 h8   __attribute__((ext_vector_type(8)));
typedef _Float16 v16h __attribute__((ext_vector_type(16)));
typedef float    v8f  __attribute__((ext_vector_type(8)));

#define H_IMG 512
#define W_IMG 512
#define TR 64                 // output tile rows per block
#define TC 32                 // output tile cols per block
#define IN_ROWS 80            // TR + 16 (5 halo + 11 extra for K padding)
#define IN_COLS 48            // TC + 16
#define IN_STRIDE 56          // f16 stride, multiple of 8 -> 16B aligned rows
#define HT_STRIDE 88          // transposed intermediate: row-dim stride (>=80, mult of 8)
#define NPLANES 48            // 16 * 3
#define TILES_R (H_IMG / TR)  // 8
#define TILES_C (W_IMG / TC)  // 16
#define BLOCKS_PER_PLANE (TILES_R * TILES_C)  // 128
#define NBLOCKS (NPLANES * BLOCKS_PER_PLANE)  // 6144
#define C1_SSIM 1e-4f
#define C2_SSIM 9e-4f
#define EPS_SSIM 1e-12f
#define INV_N (1.0f / 12582912.0f)            // 1 / (16*3*512*512)

__device__ __forceinline__ int reflect_idx(int i, int n) {
  if (i < 0) i = -i;
  if (i >= n) i = 2 * n - 2 - i;
  return i;
}

// Normalized gaussian tap: g[d] = exp(-(d-5)^2 / (2*1.5^2)) / S, S = 3.7592324
__device__ __forceinline__ float gvalf(int d) {
  if (d < 0 || d > 10) return 0.0f;
  float t = (float)(d - 5);
  return expf(t * t * (-1.0f / 4.5f)) * 0.26601148f;
}

__global__ __launch_bounds__(256) void ssim_main(const float* __restrict__ x,
                                                 const float* __restrict__ y,
                                                 float* __restrict__ partial) {
  __shared__ _Float16 sx[IN_ROWS * IN_STRIDE];
  __shared__ _Float16 sy[IN_ROWS * IN_STRIDE];
  __shared__ _Float16 ht[5][TC * HT_STRIDE];   // transposed H-blur: [signal][col][row]
  __shared__ float red[8];

  const int tid  = threadIdx.x;
  const int lane = tid & 31;
  const int wave = tid >> 5;
  const int bid  = blockIdx.x;

  const int plane = bid / BLOCKS_PER_PLANE;
  const int tile  = bid % BLOCKS_PER_PLANE;
  const int trow  = tile / TILES_C;
  const int tcol  = tile % TILES_C;
  const int r0 = trow * TR;
  const int c0 = tcol * TC;

  const float* __restrict__ xp = x + (size_t)plane * (H_IMG * W_IMG);
  const float* __restrict__ yp = y + (size_t)plane * (H_IMG * W_IMG);

  // ---- Phase A: stage reflect-padded tile (f32 -> f16) into LDS ----
  for (int idx = tid; idx < IN_ROWS * IN_COLS; idx += 256) {
    int lr = idx / IN_COLS;
    int lc = idx - lr * IN_COLS;
    int gr = reflect_idx(r0 - 5 + lr, H_IMG);
    int gc = reflect_idx(c0 - 5 + lc, W_IMG);
    sx[lr * IN_STRIDE + lc] = (_Float16)xp[gr * W_IMG + gc];
    sy[lr * IN_STRIDE + lc] = (_Float16)yp[gr * W_IMG + gc];
  }
  __syncthreads();

  const int m  = lane & 15;     // row (A) or col (B/C) index owned by this lane
  const int hs = lane >> 4;     // half-select (K-group)

  const v8f vzero = {0.f, 0.f, 0.f, 0.f, 0.f, 0.f, 0.f, 0.f};

  // ---- Stage-1 weight operand: B (32x16), B[k,n] = g[k-n], lane n = m ----
  v16h bw;
#pragma unroll
  for (int e = 0; e < 16; ++e) {
    int K = (e < 8 ? e : e + 8) + (hs ? 8 : 0);  // f16 B layout: lanes<16 K in {0..7,16..23}, lanes>=16 K in {8..15,24..31}
    bw[e] = (_Float16)gvalf(K - m);
  }

  // ---- Stage-1: horizontal blur via WMMA; 10 tile positions over 8 waves ----
  for (int pos = wave; pos < 10; pos += 8) {
    int t = pos >> 1;                    // row-tile 0..4
    int u = pos & 1;                     // col-group 0..1
    int lrBase = 16 * t;
    int lcBase = 16 * u;
    int K0a = hs ? 8 : 0;                // A f16 layout: lanes<16 K in {0..7,16..23}
    int K0b = K0a + 16;                  //               lanes>=16 K in {8..15,24..31}
    const _Float16* rowx = sx + (lrBase + m) * IN_STRIDE + lcBase;
    const _Float16* rowy = sy + (lrBase + m) * IN_STRIDE + lcBase;
    h8 x0 = *(const h8*)(rowx + K0a);
    h8 x1 = *(const h8*)(rowx + K0b);
    h8 y0 = *(const h8*)(rowy + K0a);
    h8 y1 = *(const h8*)(rowy + K0b);
    h8 xx0 = x0 * x0, xx1 = x1 * x1;     // packed f16 products
    h8 yy0 = y0 * y0, yy1 = y1 * y1;
    h8 xy0 = x0 * y0, xy1 = x1 * y1;

    v16h a[5];
    a[0] = __builtin_shufflevector(x0,  x1,  0,1,2,3,4,5,6,7,8,9,10,11,12,13,14,15);
    a[1] = __builtin_shufflevector(y0,  y1,  0,1,2,3,4,5,6,7,8,9,10,11,12,13,14,15);
    a[2] = __builtin_shufflevector(xx0, xx1, 0,1,2,3,4,5,6,7,8,9,10,11,12,13,14,15);
    a[3] = __builtin_shufflevector(yy0, yy1, 0,1,2,3,4,5,6,7,8,9,10,11,12,13,14,15);
    a[4] = __builtin_shufflevector(xy0, xy1, 0,1,2,3,4,5,6,7,8,9,10,11,12,13,14,15);

    int colLocal = lcBase + m;           // D: N = lane&15
    int rowOff   = lrBase + (hs ? 8 : 0);// D: M = v + 8*hs, v=0..7 contiguous rows
#pragma unroll
    for (int s = 0; s < 5; ++s) {
      v8f d = __builtin_amdgcn_wmma_f32_16x16x32_f16(false, a[s], false, bw,
                                                     (short)0, vzero, false, false);
      h8 hd;
#pragma unroll
      for (int v = 0; v < 8; ++v) hd[v] = (_Float16)d[v];
      *(h8*)&ht[s][colLocal * HT_STRIDE + rowOff] = hd;   // transposed store
    }
  }
  __syncthreads();

  // ---- Stage-2 weight operand: A (16x32), A[m,k] = g[k-m] ----
  v16h aw;
#pragma unroll
  for (int e = 0; e < 16; ++e) {
    int K = e + (hs ? 16 : 0);  // A in contiguous k order (same as bb): lanes<16 K=0..15, lanes>=16 K=16..31
    aw[e] = (_Float16)gvalf(K - m);
  }

  // ---- Stage-2: vertical blur via WMMA + SSIM map; 8 positions, 1 per wave ----
  float lsum = 0.0f;
  {
    int pos = wave;
    int t = pos >> 1;                    // output row-group 0..3
    int u = pos & 1;                     // col-group 0..1
    int rowBase  = 16 * t + (hs ? 16 : 0);  // B, same contiguous k as aw: lanes<16 K=0..15, lanes>=16 K=16..31
    int colLocal = 16 * u + m;              // B: N = lane&15
    v8f acc[5];
#pragma unroll
    for (int s = 0; s < 5; ++s) {
      const _Float16* p = &ht[s][colLocal * HT_STRIDE + rowBase];
      h8 b0 = *(const h8*)(p);
      h8 b1 = *(const h8*)(p + 8);
      v16h bb = __builtin_shufflevector(b0, b1, 0,1,2,3,4,5,6,7,8,9,10,11,12,13,14,15);
      acc[s] = __builtin_amdgcn_wmma_f32_16x16x32_f16(false, aw, false, bb,
                                                      (short)0, vzero, false, false);
    }
#pragma unroll
    for (int v = 0; v < 8; ++v) {
      float mu1 = acc[0][v], mu2 = acc[1][v];
      float bxx = acc[2][v], byy = acc[3][v], bxy = acc[4][v];
      float mu1s = mu1 * mu1, mu2s = mu2 * mu2, mu12 = mu1 * mu2;
      float s1  = bxx - mu1s;
      float s2  = byy - mu2s;
      float s12 = bxy - mu12;
      float num = (2.0f * mu12 + C1_SSIM) * (2.0f * s12 + C2_SSIM);
      float den = (mu1s + mu2s + C1_SSIM) * (s1 + s2 + C2_SSIM);
      float ssim = num / (den + EPS_SSIM);
      float l = 0.5f * (1.0f - ssim);
      l = fminf(fmaxf(l, 0.0f), 1.0f);
      lsum += l;
    }
  }

  // ---- Deterministic block reduction ----
#pragma unroll
  for (int off = 16; off > 0; off >>= 1) lsum += __shfl_down(lsum, off, 32);
  if (lane == 0) red[wave] = lsum;
  __syncthreads();
  if (tid == 0) {
    float s = 0.0f;
#pragma unroll
    for (int w = 0; w < 8; ++w) s += red[w];
    partial[bid] = s;
  }
}

__global__ __launch_bounds__(256) void ssim_reduce(const float* __restrict__ partial,
                                                   float* __restrict__ out, int n) {
  __shared__ float red[8];
  float s = 0.0f;
  for (int i = threadIdx.x; i < n; i += 256) s += partial[i];
#pragma unroll
  for (int off = 16; off > 0; off >>= 1) s += __shfl_down(s, off, 32);
  int lane = threadIdx.x & 31, wave = threadIdx.x >> 5;
  if (lane == 0) red[wave] = s;
  __syncthreads();
  if (threadIdx.x == 0) {
    float t = 0.0f;
#pragma unroll
    for (int w = 0; w < 8; ++w) t += red[w];
    out[0] = t * INV_N;
  }
}

extern "C" void kernel_launch(void* const* d_in, const int* in_sizes, int n_in,
                              void* d_out, int out_size, void* d_ws, size_t ws_size,
                              hipStream_t stream) {
  (void)in_sizes; (void)n_in; (void)out_size; (void)ws_size;
  const float* x = (const float*)d_in[0];
  const float* y = (const float*)d_in[1];
  float* partial = (float*)d_ws;            // NBLOCKS floats = 24 KB scratch
  ssim_main<<<dim3(NBLOCKS), dim3(256), 0, stream>>>(x, y, partial);
  ssim_reduce<<<dim3(1), dim3(256), 0, stream>>>(partial, (float*)d_out, NBLOCKS);
}
